// Wcompute_85126251807218
// MI455X (gfx1250) — matrix operand order repaired and checked by reading, not compiled, not run on hardware
//
#include <hip/hip_runtime.h>
#include <hip/hip_bf16.h>

typedef __bf16 bf16_t;
typedef __attribute__((ext_vector_type(8)))  __bf16 v8bf;
typedef __attribute__((ext_vector_type(16))) __bf16 v16bf;
typedef __attribute__((ext_vector_type(8)))  float  v8f;

#define EPS 1e-5f

// ---------------------------------------------------------------------------
// Pack fp32 weights (Cout x Cin, row-major) into bf16 with K padded to ldk.
// ---------------------------------------------------------------------------
__global__ void pack_w_kernel(const float* __restrict__ W, bf16_t* __restrict__ dst,
                              int cin, int ldk, int total) {
    int idx = blockIdx.x * blockDim.x + threadIdx.x;
    if (idx >= total) return;
    int o = idx / ldk;
    int k = idx % ldk;
    dst[idx] = (bf16_t)(k < cin ? W[o * cin + k] : 0.f);
}

// ---------------------------------------------------------------------------
// Edge features: t[b,i,j,c] = |x[b,j,c] - x[b,i,c]|, bf16, K padded 133 -> 160
// One block per edge row, 160 threads.
// ---------------------------------------------------------------------------
__global__ __launch_bounds__(160)
void edge_abs_kernel(const float* __restrict__ x, bf16_t* __restrict__ T) {
    const int row = blockIdx.x;          // b*676 + i*26 + j
    const int c   = threadIdx.x;         // 0..159
    const int b   = row / 676;
    const int r   = row % 676;
    const int i   = r / 26;
    const int j   = r % 26;
    float v = 0.f;
    if (c < 133) {
        float xj = x[((size_t)b * 26 + j) * 133 + c];
        float xi = x[((size_t)b * 26 + i) * 133 + c];
        v = fabsf(xj - xi);
    }
    T[(size_t)row * 160 + c] = (bf16_t)v;
}

// ---------------------------------------------------------------------------
// bf16 WMMA GEMM: Y[M x COUT] = A[M x lda] * W^T  (W packed as COUT x lda bf16)
// Fused per-channel sum / sum-of-squares for batch-norm statistics.
// Block = 8 waves = WM x WN wave grid; each wave owns a 16-row strip and
// NT consecutive 16-col tiles (A-fragment reuse across NT WMMAs).
// ---------------------------------------------------------------------------
template <int COUT, int WM, int WN, int NT>
__global__ __launch_bounds__(256)
void gemm_bn_stats_kernel(const bf16_t* __restrict__ A, int lda,
                          const bf16_t* __restrict__ Bw,
                          float* __restrict__ Y,
                          float* __restrict__ stats) {
    __shared__ float lsum[2 * COUT];
    const int tid = threadIdx.x;
    for (int i = tid; i < 2 * COUT; i += 256) lsum[i] = 0.f;
    __syncthreads();

    const int lane = tid & 31;
    const int wave = tid >> 5;
    const int wm   = wave % WM;
    const int wn   = wave / WM;
    const int rowBase = blockIdx.x * (16 * WM) + wm * 16;
    const int colBase = wn * (NT * 16);
    const int lhi = lane >> 4;           // 0: lanes 0-15, 1: lanes 16-31
    const int ll  = lane & 15;

    v8f acc[NT];
#pragma unroll
    for (int t = 0; t < NT; ++t)
#pragma unroll
        for (int e = 0; e < 8; ++e) acc[t][e] = 0.f;

    const bf16_t* arow = A + (size_t)(rowBase + ll) * lda;
    for (int k0 = 0; k0 < lda; k0 += 32) {
        // A fragment: contiguous K order (k = 16h + s), same as B.
        // lanes 0-15 hold K {0..15}; lanes 16-31 hold K {16..31}
        union { v16bf v; v8bf h[2]; } a;
        a.h[0] = *(const v8bf*)(arow + k0 + lhi * 16);
        a.h[1] = *(const v8bf*)(arow + k0 + lhi * 16 + 8);
#pragma unroll
        for (int t = 0; t < NT; ++t) {
            // B fragment: col = output channel; lanes 0-15 K=k0..k0+15,
            // lanes 16-31 K=k0+16..k0+31; 16 contiguous bf16 per lane.
            const bf16_t* brow =
                Bw + (size_t)(colBase + t * 16 + ll) * lda + k0 + lhi * 16;
            v16bf b = *(const v16bf*)brow;
            acc[t] = __builtin_amdgcn_wmma_f32_16x16x32_bf16(
                false, a.v, false, b, (short)0, acc[t], false, false);
        }
    }

    // Writeback + per-channel stats. C layout: VGPR r -> row r (lanes 0-15)
    // or row r+8 (lanes 16-31); col = lane&15.
#pragma unroll
    for (int t = 0; t < NT; ++t) {
        const int col = colBase + t * 16 + ll;
        float s = 0.f, ss = 0.f;
#pragma unroll
        for (int r = 0; r < 8; ++r) {
            const int row = rowBase + r + lhi * 8;
            float v = acc[t][r];
            Y[(size_t)row * COUT + col] = v;
            s += v;
            ss += v * v;
        }
        atomicAdd(&lsum[col], s);
        atomicAdd(&lsum[COUT + col], ss);
    }
    __syncthreads();
    for (int i = tid; i < 2 * COUT; i += 256) atomicAdd(&stats[i], lsum[i]);
}

// ---------------------------------------------------------------------------
// stats -> (scale, shift) per channel.  Conv bias cancels through BN.
// ---------------------------------------------------------------------------
__global__ void finalize_stats_kernel(const float* __restrict__ stats,
                                      const float* __restrict__ g,
                                      const float* __restrict__ be,
                                      float* __restrict__ coef,
                                      int cout, float invM) {
    int ch = threadIdx.x;
    if (ch >= cout) return;
    float mean = stats[ch] * invM;
    float var  = stats[cout + ch] * invM - mean * mean;
    float sc   = g[ch] * rsqrtf(var + EPS);
    coef[2 * ch]     = sc;
    coef[2 * ch + 1] = be[ch] - mean * sc;
}

__global__ void zero_f32_kernel(float* p, int n) {
    int i = blockIdx.x * blockDim.x + threadIdx.x;
    if (i < n) p[i] = 0.f;
}

// ---------------------------------------------------------------------------
// BN (scale/shift) + leaky-relu(0.01), fp32 -> bf16
// ---------------------------------------------------------------------------
__global__ void apply_bn_lrelu_kernel(const float* __restrict__ Y,
                                      const float* __restrict__ coef,
                                      bf16_t* __restrict__ out,
                                      int cout, size_t total) {
    size_t idx = (size_t)blockIdx.x * blockDim.x + threadIdx.x;
    if (idx >= total) return;
    int ch = (int)(idx % (size_t)cout);
    float v = Y[idx] * coef[2 * ch] + coef[2 * ch + 1];
    v = v > 0.f ? v : 0.01f * v;
    out[idx] = (bf16_t)v;
}

// ---------------------------------------------------------------------------
// Final 96->1 linear, w1 = score - w0; write w_cat = [w0, w1] and score buf
// ---------------------------------------------------------------------------
__global__ void score_kernel(const bf16_t* __restrict__ act,
                             const float* __restrict__ Wl,
                             const float* __restrict__ bl,
                             const float* __restrict__ Wid,
                             float* __restrict__ outW,
                             float* __restrict__ sc, int M) {
    int row = blockIdx.x * blockDim.x + threadIdx.x;
    if (row >= M) return;
    const bf16_t* a = act + (size_t)row * 96;
    float s = 0.f;
#pragma unroll 8
    for (int c = 0; c < 96; ++c) s += (float)a[c] * Wl[c];
    s += bl[0];
    float w0 = Wid[row];
    float w1 = s - w0;
    outW[2 * row]     = w0;
    outW[2 * row + 1] = w1;
    sc[row] = w1;
}

// ---------------------------------------------------------------------------
// Per-(b,i): softmax over j of w1, then m0/m1 weighted neighbor sums -> h
// ---------------------------------------------------------------------------
__global__ __launch_bounds__(160)
void combine_kernel(const float* __restrict__ x,
                    const float* __restrict__ Wid,
                    const float* __restrict__ sc,
                    float* __restrict__ outH) {
    __shared__ float p0[26];
    __shared__ float p1[26];
    const int bi  = blockIdx.x;          // b*26 + i
    const int b   = bi / 26;
    const int tid = threadIdx.x;
    if (tid < 26) {
        p0[tid] = Wid[(size_t)bi * 26 + tid];
        p1[tid] = sc[(size_t)bi * 26 + tid];
    }
    __syncthreads();
    if (tid == 0) {
        float mx = -1e30f;
        for (int j = 0; j < 26; ++j) mx = fmaxf(mx, p1[j]);
        float sum = 0.f;
        for (int j = 0; j < 26; ++j) { float e = __expf(p1[j] - mx); p1[j] = e; sum += e; }
        float inv = 1.f / sum;
        for (int j = 0; j < 26; ++j) p1[j] *= inv;
    }
    __syncthreads();
    if (tid < 133) {
        float m0 = 0.f, m1 = 0.f;
        const float* xb = x + (size_t)b * 26 * 133 + tid;
        for (int j = 0; j < 26; ++j) {
            float xv = xb[(size_t)j * 133];
            m0 += p0[j] * xv;
            m1 += p1[j] * xv;
        }
        outH[(size_t)bi * 266 + tid]       = m0;
        outH[(size_t)bi * 266 + 133 + tid] = m1;
    }
}

// ---------------------------------------------------------------------------
extern "C" void kernel_launch(void* const* d_in, const int* in_sizes, int n_in,
                              void* d_out, int out_size, void* d_ws, size_t ws_size,
                              hipStream_t stream) {
    (void)in_sizes; (void)n_in; (void)out_size; (void)ws_size;
    const int    BS = 512, N = 26;
    const size_t M  = (size_t)BS * N * N;         // 346112 (divisible by 64)
    const float  invM = 1.f / (float)M;

    const float* x   = (const float*)d_in[0];
    const float* Wid = (const float*)d_in[1];
    const float* W1  = (const float*)d_in[2];
    const float* g1  = (const float*)d_in[4];
    const float* be1 = (const float*)d_in[5];
    const float* W2  = (const float*)d_in[6];
    const float* g2  = (const float*)d_in[8];
    const float* be2 = (const float*)d_in[9];
    const float* W3  = (const float*)d_in[10];
    const float* g3  = (const float*)d_in[12];
    const float* be3 = (const float*)d_in[13];
    const float* W4  = (const float*)d_in[14];
    const float* g4  = (const float*)d_in[16];
    const float* be4 = (const float*)d_in[17];
    const float* Wl  = (const float*)d_in[18];
    const float* bl  = (const float*)d_in[19];

    // workspace carve-up (256B aligned)
    char*  ws  = (char*)d_ws;
    size_t off = 0;
    auto carve = [&](size_t bytes) -> void* {
        void* p = ws + off;
        off += (bytes + 255) & ~(size_t)255;
        return p;
    };
    bf16_t* actA  = (bf16_t*)carve(M * 192 * sizeof(bf16_t));
    bf16_t* actB  = (bf16_t*)carve(M * 192 * sizeof(bf16_t));
    float*  Y     = (float*) carve(M * 192 * sizeof(float));
    bf16_t* wp1   = (bf16_t*)carve(192 * 160 * sizeof(bf16_t));
    bf16_t* wp2   = (bf16_t*)carve(192 * 192 * sizeof(bf16_t));
    bf16_t* wp3   = (bf16_t*)carve( 96 * 192 * sizeof(bf16_t));
    bf16_t* wp4   = (bf16_t*)carve( 96 *  96 * sizeof(bf16_t));
    float*  stats = (float*) carve(384 * sizeof(float));
    float*  coef  = (float*) carve(384 * sizeof(float));
    float*  sc    = (float*) carve(M * sizeof(float));

    // pack weights to bf16 (K padded to 32-multiple)
    pack_w_kernel<<<(192 * 160 + 255) / 256, 256, 0, stream>>>(W1, wp1, 133, 160, 192 * 160);
    pack_w_kernel<<<(192 * 192 + 255) / 256, 256, 0, stream>>>(W2, wp2, 192, 192, 192 * 192);
    pack_w_kernel<<<( 96 * 192 + 255) / 256, 256, 0, stream>>>(W3, wp3, 192, 192,  96 * 192);
    pack_w_kernel<<<( 96 *  96 + 255) / 256, 256, 0, stream>>>(W4, wp4,  96,  96,  96 *  96);

    // edge features -> actA (lda = 160)
    edge_abs_kernel<<<(int)M, 160, 0, stream>>>(x, actA);

    const int gemmBlocks192 = (int)(M / 32);   // block covers 32 rows x 192 cols
    const int gemmBlocks96  = (int)(M / 64);   // block covers 64 rows x  96 cols
    const int applyBlocks192 = (int)((M * 192 + 255) / 256);
    const int applyBlocks96  = (int)((M * 96 + 255) / 256);

    // layer 1: 160 -> 192
    zero_f32_kernel<<<2, 256, 0, stream>>>(stats, 384);
    gemm_bn_stats_kernel<192, 2, 4, 3><<<gemmBlocks192, 256, 0, stream>>>(actA, 160, wp1, Y, stats);
    finalize_stats_kernel<<<1, 192, 0, stream>>>(stats, g1, be1, coef, 192, invM);
    apply_bn_lrelu_kernel<<<applyBlocks192, 256, 0, stream>>>(Y, coef, actB, 192, M * 192);

    // layer 2: 192 -> 192
    zero_f32_kernel<<<2, 256, 0, stream>>>(stats, 384);
    gemm_bn_stats_kernel<192, 2, 4, 3><<<gemmBlocks192, 256, 0, stream>>>(actB, 192, wp2, Y, stats);
    finalize_stats_kernel<<<1, 192, 0, stream>>>(stats, g2, be2, coef, 192, invM);
    apply_bn_lrelu_kernel<<<applyBlocks192, 256, 0, stream>>>(Y, coef, actA, 192, M * 192);

    // layer 3: 192 -> 96
    zero_f32_kernel<<<2, 256, 0, stream>>>(stats, 384);
    gemm_bn_stats_kernel<96, 4, 2, 3><<<gemmBlocks96, 256, 0, stream>>>(actA, 192, wp3, Y, stats);
    finalize_stats_kernel<<<1, 96, 0, stream>>>(stats, g3, be3, coef, 96, invM);
    apply_bn_lrelu_kernel<<<applyBlocks96, 256, 0, stream>>>(Y, coef, actB, 96, M * 96);

    // layer 4: 96 -> 96
    zero_f32_kernel<<<2, 256, 0, stream>>>(stats, 384);
    gemm_bn_stats_kernel<96, 4, 2, 3><<<gemmBlocks96, 256, 0, stream>>>(actB, 96, wp4, Y, stats);
    finalize_stats_kernel<<<1, 96, 0, stream>>>(stats, g4, be4, coef, 96, invM);
    apply_bn_lrelu_kernel<<<applyBlocks96, 256, 0, stream>>>(Y, coef, actA, 96, M * 96);

    // outputs: w_cat (M*2 floats) then h (BS*N*266 floats)
    float* outW = (float*)d_out;
    float* outH = outW + M * 2;
    score_kernel<<<(int)((M + 255) / 256), 256, 0, stream>>>(actA, Wl, bl, Wid, outW, sc, (int)M);
    combine_kernel<<<BS * N, 160, 0, stream>>>(x, Wid, sc, outH);
}
